// CapsuleLayer_70574902608325
// MI455X (gfx1250) — matrix operand order repaired and checked by reading, not compiled, not run on hardware
//
#include <hip/hip_runtime.h>
#include <hip/hip_bf16.h>

// ---- problem constants --------------------------------------------------
#define BB   32            // batch
#define NC   32            // capsules
#define OC   32            // out caps dim
#define IC   256           // in channels
#define HH   71            // input H = W
#define KER  9
#define HP   32            // output H' = W'
#define MROWS (NC*OC)      // 1024  GEMM M
#define KTOT  (IC*KER*KER) // 20736 GEMM K
#define KSTEPS (KTOT/32)   // 648

typedef __attribute__((ext_vector_type(16))) __bf16 v16bf;
typedef __attribute__((ext_vector_type(8)))  float  v8f;
typedef __attribute__((ext_vector_type(4)))  float  v4f;

// fp32 -> bf16, round-to-nearest-even (branch-free, compile-safe)
static __device__ __forceinline__ __bf16 tobf(float f) {
    unsigned u = __builtin_bit_cast(unsigned, f);
    u += 0x7fffu + ((u >> 16) & 1u);
    unsigned short h = (unsigned short)(u >> 16);
    return __builtin_bit_cast(__bf16, h);
}

// ---- LDS layout (bytes), dynamic shared ---------------------------------
#define U_STR       33                        // u_hat row stride (floats)
#define U_SLICE_F   (MROWS * U_STR)           // 33792 floats per slice
#define U_BYTES     (2 * U_SLICE_F * 4)       // 270336 (two b-slices)
#define BT_STR      40                        // bf16 per staged row
#define BT_OFF      U_BYTES                   // two B tiles [32 n][40 k] bf16
#define BT_TILE_B   (32 * BT_STR * 2)         // 2560
#define BIJ_OFF     (BT_OFF + 2 * BT_TILE_B)
#define CB_OFF      (BIJ_OFF + MROWS * 4)
#define SB_OFF      (CB_OFF + MROWS * 4)
#define VB_OFF      (SB_OFF + 32 * 33 * 4)
#define SUM_OFF     (VB_OFF + 32 * 33 * 4)
#define FAC_OFF     (SUM_OFF + 32 * 4)
#define SMEM_BYTES  (FAC_OFF + 32 * 4)        // ~292 KB < 320 KB WGP LDS

// =========================================================================
// Pre-pass: fp32 W -> bf16, fragment-ordered for the WMMA A operand.
//   group (kk, M, half) -> 16 contiguous bf16:
//   K = kk*32 + half*16 + i
// =========================================================================
__global__ void convert_w_swz(const float* __restrict__ Wf,
                              __bf16* __restrict__ Wswz) {
    size_t t = (size_t)blockIdx.x * 256 + threadIdx.x;  // 0 .. KSTEPS*1024*2-1
    int half = (int)(t & 1);
    size_t rest = t >> 1;
    int M  = (int)(rest & (MROWS - 1));
    int kk = (int)(rest >> 10);
    const float* src = Wf + (size_t)M * KTOT + kk * 32 + half * 16;
    v4f a0 = *(const v4f*)(src);
    v4f a1 = *(const v4f*)(src + 4);
    v4f a2 = *(const v4f*)(src + 8);
    v4f a3 = *(const v4f*)(src + 12);
    v16bf o;
#pragma unroll
    for (int j = 0; j < 4; ++j) {
        o[j]      = tobf(a0[j]);
        o[4 + j]  = tobf(a1[j]);
        o[8 + j]  = tobf(a2[j]);
        o[12 + j] = tobf(a3[j]);
    }
    *(v16bf*)(Wswz + t * 16) = o;   // t*16 == ((kk*1024+M)*2+half)*16
}

// =========================================================================
// Fused conv (implicit GEMM, bf16 WMMA) + dynamic routing.
// Workgroup = 1024 threads (32 wave32s) handles (hp, b0=2bp, b1=2bp+1).
// Each wave owns 32 M-rows (2 mtiles); each A fragment feeds 4 WMMAs
// (2 b-slices x 2 n-tiles) -> halved L2 weight traffic, 64 acc VGPRs/lane.
// =========================================================================
__global__ void capsule_fused_wmma(const float* __restrict__ x,
                                   const __bf16* __restrict__ Wswz,
                                   const float* __restrict__ bias,
                                   float* __restrict__ out) {
    extern __shared__ char smem[];
    float*  U0   = (float*)smem;                          // [1024][33] slice 0
    float*  U1   = U0 + U_SLICE_F;                        // slice 1
    __bf16* Bt0  = (__bf16*)(smem + BT_OFF);              // [32 n][40 k]
    __bf16* Bt1  = (__bf16*)(smem + BT_OFF + BT_TILE_B);
    float*  bij  = (float*)(smem + BIJ_OFF);
    float*  cb   = (float*)(smem + CB_OFF);
    float*  sb   = (float*)(smem + SB_OFF);
    float*  vb   = (float*)(smem + VB_OFF);
    float*  sums = (float*)(smem + SUM_OFF);
    float*  fac  = (float*)(smem + FAC_OFF);

    const int hp  = blockIdx.x;             // 0..31
    const int bp  = blockIdx.y;             // 0..15  -> batches 2bp, 2bp+1
    const int tid = threadIdx.x;            // 0..1023 (32 wave32s)
    const int lane = tid & 31;
    const int wave = tid >> 5;
    const int mbase = wave << 5;            // 32 M-rows per wave
    const int lrow  = lane & 15;
    const int halfsel = lane >> 4;

    const float* xb0 = x + (size_t)(2 * bp)     * IC * HH * HH;
    const float* xb1 = x + (size_t)(2 * bp + 1) * IC * HH * HH;

    // ---------------- implicit-GEMM conv --------------------------------
    v8f acc[2][4];                            // [mt][slice*2 + nt]
    const v8f vzero = {0.f,0.f,0.f,0.f,0.f,0.f,0.f,0.f};
#pragma unroll
    for (int mt = 0; mt < 2; ++mt)
#pragma unroll
        for (int j = 0; j < 4; ++j) acc[mt][j] = vzero;

    // this thread's fixed staging coordinates (1 element per slice per iter)
    const int s_wp = tid >> 5, s_kl = tid & 31;

    for (int kk = 0; kk < KSTEPS; ++kk) {
        const int kbase = kk * 32;

        // stage both 32x32 im2col patches as bf16, K-transposed
        {
            int k  = kbase + s_kl;
            int c  = k / 81;  int r = k - c * 81;
            int kh = r / 9;   int kw = r - kh * 9;
            size_t off = ((size_t)c * HH + (hp * 2 + kh)) * HH + (s_wp * 2 + kw);
            Bt0[s_wp * BT_STR + s_kl] = tobf(xb0[off]);
            Bt1[s_wp * BT_STR + s_kl] = tobf(xb1[off]);
        }
        __syncthreads();

        // B fragments: lane's v16bf = contiguous K-run [half*16..+15] at col n
        v16bf bfrag[4];
#pragma unroll
        for (int nt = 0; nt < 2; ++nt) {
            int n = nt * 16 + lrow;
            bfrag[nt]     = *(const v16bf*)(Bt0 + n * BT_STR + halfsel * 16);
            bfrag[2 + nt] = *(const v16bf*)(Bt1 + n * BT_STR + halfsel * 16);
        }

        // A fragments: one contiguous 32B load feeds 4 WMMAs
#pragma unroll
        for (int mt = 0; mt < 2; ++mt) {
            int M = mbase + mt * 16 + lrow;
            const __bf16* ap =
                Wswz + ((size_t)kk * MROWS + M) * 32 + halfsel * 16;
            __builtin_prefetch((const void*)(ap + (size_t)MROWS * 32), 0, 1);
            v16bf af = *(const v16bf*)ap;
            acc[mt][0] = __builtin_amdgcn_wmma_f32_16x16x32_bf16(
                false, af, false, bfrag[0], (short)0, acc[mt][0], false, false);
            acc[mt][1] = __builtin_amdgcn_wmma_f32_16x16x32_bf16(
                false, af, false, bfrag[1], (short)0, acc[mt][1], false, false);
            acc[mt][2] = __builtin_amdgcn_wmma_f32_16x16x32_bf16(
                false, af, false, bfrag[2], (short)0, acc[mt][2], false, false);
            acc[mt][3] = __builtin_amdgcn_wmma_f32_16x16x32_bf16(
                false, af, false, bfrag[3], (short)0, acc[mt][3], false, false);
        }
        __syncthreads();
    }

    // dump accumulators (+bias) into the two LDS u_hat tiles
    // C/D layout: VGPR r -> lanes 0-15: M=r ; lanes 16-31: M=8+r ; N=lane&15
#pragma unroll
    for (int mt = 0; mt < 2; ++mt)
#pragma unroll
        for (int j = 0; j < 4; ++j) {
            float* Us = (j < 2) ? U0 : U1;
            int N = (j & 1) * 16 + lrow;
#pragma unroll
            for (int r = 0; r < 8; ++r) {
                int M = mbase + mt * 16 + r + (halfsel << 3);
                Us[M * U_STR + N] = acc[mt][j][r] + bias[M];
            }
        }
    __syncthreads();

    // ------------- dynamic routing, per slice, all in LDS ----------------
    // 1024 threads <-> 1024 (nc,oc) pairs / (nc,wp) cells, 1:1.
    const int p_nc = tid >> 5;              // nc for pair p=tid
    const int p_lo = tid & 31;              // oc (pair view) / wp (cell view)

    for (int s = 0; s < 2; ++s) {
        float* Ulds = (s == 0) ? U0 : U1;
        const int b = 2 * bp + s;

        bij[tid] = 0.0f;
        __syncthreads();

        for (int it = 0; it < 3; ++it) {
            cb[tid] = __expf(bij[tid]);
            __syncthreads();
            if (tid < 32) {                      // softmax over nc per oc
                float sum = 0.f;
                for (int nc = 0; nc < NC; ++nc) sum += cb[nc * 32 + tid];
                sums[tid] = 1.f / sum;
            }
            __syncthreads();
            cb[tid] *= sums[p_lo];
            __syncthreads();
            {                                    // s = sum_oc c * u_hat
                float a = 0.f;
                for (int oc = 0; oc < OC; ++oc)
                    a += cb[p_nc * 32 + oc] * Ulds[(p_nc * 32 + oc) * U_STR + p_lo];
                sb[p_nc * 33 + p_lo] = a;
            }
            __syncthreads();
            if (tid < 32) {                      // squash factor over wp
                float n2 = 0.f;
                for (int wp = 0; wp < 32; ++wp) {
                    float v = sb[tid * 33 + wp]; n2 += v * v;
                }
                float n = __fsqrt_rn(n2);
                fac[tid] = n / (1.f + n2);
            }
            __syncthreads();
            vb[p_nc * 33 + p_lo] = fac[p_nc] * sb[p_nc * 33 + p_lo];
            __syncthreads();
            if (it < 2) {                        // b_ij += <u_hat, v>
                float ag = 0.f;
                for (int wp = 0; wp < 32; ++wp)
                    ag += Ulds[tid * U_STR + wp] * vb[p_nc * 33 + wp];
                bij[tid] += ag;
                __syncthreads();
            }
        }

        // write v: out[b][nc][0][hp][wp]
        out[(((size_t)b * NC + p_nc) * HP + hp) * HP + p_lo] = vb[p_nc * 33 + p_lo];
        __syncthreads();   // before slice 1 reuses bij/cb/sb/vb
    }
}

extern "C" void kernel_launch(void* const* d_in, const int* in_sizes, int n_in,
                              void* d_out, int out_size, void* d_ws, size_t ws_size,
                              hipStream_t stream) {
    const float* x    = (const float*)d_in[0];
    const float* W    = (const float*)d_in[1];
    const float* bias = (const float*)d_in[2];
    float* out        = (float*)d_out;
    __bf16* Wswz      = (__bf16*)d_ws;        // needs KTOT*MROWS*2 = 42.5 MB

    (void)in_sizes; (void)n_in; (void)out_size; (void)ws_size;

    // pre-pass: fragment-ordered bf16 weights (one 16-elem group per thread)
    convert_w_swz<<<dim3(KSTEPS * 8), dim3(256), 0, stream>>>(W, Wswz);

    hipFuncSetAttribute((const void*)capsule_fused_wmma,
                        hipFuncAttributeMaxDynamicSharedMemorySize, SMEM_BYTES);
    dim3 grid(HP, BB / 2);    // (hp, batch-pair) -> 512 workgroups
    dim3 block(1024);         // 32 wave32s
    capsule_fused_wmma<<<grid, block, SMEM_BYTES, stream>>>(x, Wswz, bias, out);
}
